// Decoder_5085241278799
// MI455X (gfx1250) — matrix operand order repaired and checked by reading, not compiled, not run on hardware
//
#include <hip/hip_runtime.h>

// ---------------------------------------------------------------------------
// CDNA5 (gfx1250) GRU-attention decoder.
// Sequential recurrence with small WMMA GEMMs; the dominant 231-GFLOP vocab
// projection is hoisted out of the time loop into ONE software-pipelined
// bf16 WMMA GEMM ([2016x1792] x [1792x32000], W_out bf16 = 115 MB, L2-resident).
// ---------------------------------------------------------------------------

typedef __attribute__((ext_vector_type(16))) __bf16 v16bf;
typedef __attribute__((ext_vector_type(8)))  float  v8f;

union Frag16 { v16bf v; unsigned u[8]; };

__device__ __forceinline__ unsigned short f2bf(float f) {
  unsigned u = __float_as_uint(f);
  unsigned r = (u + 0x7FFFu + ((u >> 16) & 1u)) >> 16;   // round-to-nearest-even
  return (unsigned short)r;
}

// A fragment 16x32 bf16 (ISA 7.12.2): VGPR v<4 -> K = half*8 + 2v ; v>=4 -> +16
__device__ __forceinline__ void load_a_frag(const unsigned short* __restrict__ row,
                                            int k0, int half, Frag16& f) {
#pragma unroll
  for (int v = 0; v < 4; ++v)
    f.u[v] = *(const unsigned*)(row + k0 + half * 8 + 2 * v);
#pragma unroll
  for (int v = 0; v < 4; ++v)
    f.u[4 + v] = *(const unsigned*)(row + k0 + 16 + half * 8 + 2 * v);
}

// B fragment 32x16 bf16: same K order as A: VGPR v<4 -> K = half*8 + 2v ; v>=4 -> +16
__device__ __forceinline__ void load_b_frag(const unsigned short* __restrict__ row,
                                            int k0, int half, Frag16& f) {
#pragma unroll
  for (int v = 0; v < 8; ++v)
    f.u[v] = *(const unsigned*)(row + k0 + (v >> 2) * 16 + half * 8 + 2 * (v & 3));
}

// ---------------------------------------------------------------------------
// fp32 -> bf16 conversion (grid-stride)
// ---------------------------------------------------------------------------
__global__ void convert_f32_bf16(const float* __restrict__ src,
                                 unsigned short* __restrict__ dst, long n) {
  long i = (long)blockIdx.x * blockDim.x + threadIdx.x;
  long stride = (long)gridDim.x * blockDim.x;
  for (; i < n; i += stride) dst[i] = f2bf(src[i]);
}

// ---------------------------------------------------------------------------
// Small-M bf16 WMMA GEMM: C[M,N](f32) = A[M,K]*B[N,K]^T + bias[N]
// Block 256 thr (8 waves), tile 32(M) x 256(N); wave tile 16x64.
// For the M=32 recurrence GEMMs (q, gi, gh). M%32==0, N%256==0, K%32==0.
// ---------------------------------------------------------------------------
__global__ void gemm_bf16_wmma(const unsigned short* __restrict__ A, int lda,
                               const unsigned short* __restrict__ B, int ldb,
                               const float* __restrict__ bias,
                               float* __restrict__ C, long ldc, int K) {
  const int lane = threadIdx.x & 31;
  const int wave = threadIdx.x >> 5;
  const int wm   = wave & 1;
  const int wn   = wave >> 1;
  const int half = lane >> 4;
  const int l16  = lane & 15;

  const int mBase = blockIdx.y * 32 + wm * 16;
  const int nBase = blockIdx.x * 256 + wn * 64;

  const unsigned short* aRow = A + (size_t)(mBase + l16) * lda;
  const unsigned short* bRow = B + (size_t)(nBase + l16) * ldb;

  v8f acc0 = {}, acc1 = {}, acc2 = {}, acc3 = {};
  const int ksteps = K >> 5;

  for (int kk = 0; kk < ksteps; ++kk) {
    const int k0 = kk << 5;
    Frag16 a, b0, b1, b2, b3;
    load_a_frag(aRow, k0, half, a);
    load_b_frag(bRow, k0, half, b0);
    load_b_frag(bRow + (size_t)16 * ldb, k0, half, b1);
    load_b_frag(bRow + (size_t)32 * ldb, k0, half, b2);
    load_b_frag(bRow + (size_t)48 * ldb, k0, half, b3);
    acc0 = __builtin_amdgcn_wmma_f32_16x16x32_bf16(false, a.v, false, b0.v, (short)0, acc0, false, false);
    acc1 = __builtin_amdgcn_wmma_f32_16x16x32_bf16(false, a.v, false, b1.v, (short)0, acc1, false, false);
    acc2 = __builtin_amdgcn_wmma_f32_16x16x32_bf16(false, a.v, false, b2.v, (short)0, acc2, false, false);
    acc3 = __builtin_amdgcn_wmma_f32_16x16x32_bf16(false, a.v, false, b3.v, (short)0, acc3, false, false);
  }

  const int n0 = nBase + l16;
  const float bs0 = bias ? bias[n0]      : 0.0f;
  const float bs1 = bias ? bias[n0 + 16] : 0.0f;
  const float bs2 = bias ? bias[n0 + 32] : 0.0f;
  const float bs3 = bias ? bias[n0 + 48] : 0.0f;
#pragma unroll
  for (int v = 0; v < 8; ++v) {
    const int m = mBase + half * 8 + v;
    float* cr = C + (size_t)m * ldc;
    cr[n0]      = acc0[v] + bs0;
    cr[n0 + 16] = acc1[v] + bs1;
    cr[n0 + 32] = acc2[v] + bs2;
    cr[n0 + 48] = acc3[v] + bs3;
  }
}

// ---------------------------------------------------------------------------
// Big-tile, software-pipelined bf16 WMMA GEMM.
// Block 256 thr (8 waves), block tile 64(M) x 256(N); wave tile 32x64:
//   2 A-frags x 4 B-frags -> 8 wmma per k-step (1.5 b128 loads per wmma),
//   double-buffered fragments so k-step n+1's loads overlap k-step n's wmmas.
// grid.y = Mpad/64 (Mpad%64==0); stores guarded by m < Mvalid.
// out_mode 0: C[m*ldc+n] ; out_mode 1: logits permutation m=t*32+b -> b*63+t.
// K%64==0 (even number of 32-wide k-steps).
// ---------------------------------------------------------------------------
__global__ void gemm_bf16_wmma_big(const unsigned short* __restrict__ A, int lda,
                                   const unsigned short* __restrict__ B, int ldb,
                                   const float* __restrict__ bias,
                                   float* __restrict__ C, long ldc,
                                   int K, int Mvalid, int out_mode) {
  const int lane = threadIdx.x & 31;
  const int wave = threadIdx.x >> 5;
  const int wm   = wave & 1;          // which 32-row half of the 64-row block
  const int wn   = wave >> 1;         // N quad
  const int half = lane >> 4;
  const int l16  = lane & 15;

  const int mBase = blockIdx.y * 64 + wm * 32;
  const int nBase = blockIdx.x * 256 + wn * 64;

  const unsigned short* aRow0 = A + (size_t)(mBase + l16) * lda;        // rows m..m+15
  const unsigned short* aRow1 = aRow0 + (size_t)16 * lda;               // rows m+16..m+31
  const unsigned short* bRow0 = B + (size_t)(nBase + l16) * ldb;
  const unsigned short* bRow1 = bRow0 + (size_t)16 * ldb;
  const unsigned short* bRow2 = bRow0 + (size_t)32 * ldb;
  const unsigned short* bRow3 = bRow0 + (size_t)48 * ldb;

  v8f acc00 = {}, acc01 = {}, acc02 = {}, acc03 = {};
  v8f acc10 = {}, acc11 = {}, acc12 = {}, acc13 = {};

  Frag16 aA0, aA1, bA0, bA1, bA2, bA3;   // pipeline stage A
  Frag16 aB0, aB1, bB0, bB1, bB2, bB3;   // pipeline stage B

  const int ksteps = K >> 5;             // even by contract

  // prologue: stage A <- k-step 0
  load_a_frag(aRow0, 0, half, aA0);
  load_a_frag(aRow1, 0, half, aA1);
  load_b_frag(bRow0, 0, half, bA0);
  load_b_frag(bRow1, 0, half, bA1);
  load_b_frag(bRow2, 0, half, bA2);
  load_b_frag(bRow3, 0, half, bA3);

  for (int kk = 0; kk < ksteps; kk += 2) {
    // prefetch k-step kk+1 into stage B while computing stage A
    {
      const int k1 = (kk + 1) << 5;
      load_a_frag(aRow0, k1, half, aB0);
      load_a_frag(aRow1, k1, half, aB1);
      load_b_frag(bRow0, k1, half, bB0);
      load_b_frag(bRow1, k1, half, bB1);
      load_b_frag(bRow2, k1, half, bB2);
      load_b_frag(bRow3, k1, half, bB3);
    }
    acc00 = __builtin_amdgcn_wmma_f32_16x16x32_bf16(false, aA0.v, false, bA0.v, (short)0, acc00, false, false);
    acc10 = __builtin_amdgcn_wmma_f32_16x16x32_bf16(false, aA1.v, false, bA0.v, (short)0, acc10, false, false);
    acc01 = __builtin_amdgcn_wmma_f32_16x16x32_bf16(false, aA0.v, false, bA1.v, (short)0, acc01, false, false);
    acc11 = __builtin_amdgcn_wmma_f32_16x16x32_bf16(false, aA1.v, false, bA1.v, (short)0, acc11, false, false);
    acc02 = __builtin_amdgcn_wmma_f32_16x16x32_bf16(false, aA0.v, false, bA2.v, (short)0, acc02, false, false);
    acc12 = __builtin_amdgcn_wmma_f32_16x16x32_bf16(false, aA1.v, false, bA2.v, (short)0, acc12, false, false);
    acc03 = __builtin_amdgcn_wmma_f32_16x16x32_bf16(false, aA0.v, false, bA3.v, (short)0, acc03, false, false);
    acc13 = __builtin_amdgcn_wmma_f32_16x16x32_bf16(false, aA1.v, false, bA3.v, (short)0, acc13, false, false);

    // prefetch k-step kk+2 into stage A while computing stage B
    if (kk + 2 < ksteps) {
      const int k2 = (kk + 2) << 5;
      load_a_frag(aRow0, k2, half, aA0);
      load_a_frag(aRow1, k2, half, aA1);
      load_b_frag(bRow0, k2, half, bA0);
      load_b_frag(bRow1, k2, half, bA1);
      load_b_frag(bRow2, k2, half, bA2);
      load_b_frag(bRow3, k2, half, bA3);
    }
    acc00 = __builtin_amdgcn_wmma_f32_16x16x32_bf16(false, aB0.v, false, bB0.v, (short)0, acc00, false, false);
    acc10 = __builtin_amdgcn_wmma_f32_16x16x32_bf16(false, aB1.v, false, bB0.v, (short)0, acc10, false, false);
    acc01 = __builtin_amdgcn_wmma_f32_16x16x32_bf16(false, aB0.v, false, bB1.v, (short)0, acc01, false, false);
    acc11 = __builtin_amdgcn_wmma_f32_16x16x32_bf16(false, aB1.v, false, bB1.v, (short)0, acc11, false, false);
    acc02 = __builtin_amdgcn_wmma_f32_16x16x32_bf16(false, aB0.v, false, bB2.v, (short)0, acc02, false, false);
    acc12 = __builtin_amdgcn_wmma_f32_16x16x32_bf16(false, aB1.v, false, bB2.v, (short)0, acc12, false, false);
    acc03 = __builtin_amdgcn_wmma_f32_16x16x32_bf16(false, aB0.v, false, bB3.v, (short)0, acc03, false, false);
    acc13 = __builtin_amdgcn_wmma_f32_16x16x32_bf16(false, aB1.v, false, bB3.v, (short)0, acc13, false, false);
  }

  const int n0 = nBase + l16;
  const float bs0 = bias ? bias[n0]      : 0.0f;
  const float bs1 = bias ? bias[n0 + 16] : 0.0f;
  const float bs2 = bias ? bias[n0 + 32] : 0.0f;
  const float bs3 = bias ? bias[n0 + 48] : 0.0f;
#pragma unroll
  for (int i = 0; i < 2; ++i) {
    const v8f* a0 = (i == 0) ? &acc00 : &acc10;
    const v8f* a1 = (i == 0) ? &acc01 : &acc11;
    const v8f* a2 = (i == 0) ? &acc02 : &acc12;
    const v8f* a3 = (i == 0) ? &acc03 : &acc13;
#pragma unroll
    for (int v = 0; v < 8; ++v) {
      const int m = mBase + i * 16 + half * 8 + v;
      if (m >= Mvalid) continue;
      size_t row;
      if (out_mode == 1) { const int t = m >> 5; const int b = m & 31; row = (size_t)b * 63 + t; }
      else               { row = (size_t)m; }
      float* cr = C + row * ldc;
      cr[n0]      = (*a0)[v] + bs0;
      cr[n0 + 16] = (*a1)[v] + bs1;
      cr[n0 + 32] = (*a2)[v] + bs2;
      cr[n0 + 48] = (*a3)[v] + bs3;
    }
  }
}

// ---------------------------------------------------------------------------
// Init: h_state(f32) = h0 ; x_cat h-slot = bf16(h0)
// ---------------------------------------------------------------------------
__global__ void init_state(const float* __restrict__ h0,
                           float* __restrict__ h_state,
                           unsigned short* __restrict__ x_cat) {
  int idx = blockIdx.x * blockDim.x + threadIdx.x;   // 0..16383
  if (idx >= 32 * 512) return;
  const int b = idx >> 9, j = idx & 511;
  float h = h0[idx];
  h_state[idx] = h;
  x_cat[(size_t)b * 1792 + 1280 + j] = f2bf(h);
}

// ---------------------------------------------------------------------------
// Attention step: scores = q . k_enc ; mask ; softmax ; context = attn @ enc
// Plus embedding lookup. One block per batch row b (32 blocks x 256 thr).
// ---------------------------------------------------------------------------
__global__ void attn_kernel(const int* __restrict__ tgt,
                            const int* __restrict__ mask,
                            const float* __restrict__ enc,
                            const float* __restrict__ emb_table,
                            const float* __restrict__ q,
                            const float* __restrict__ k_enc,
                            unsigned short* __restrict__ x_cat,
                            unsigned short* __restrict__ concat_bf,
                            int t) {
  const int b   = blockIdx.x;
  const int tid = threadIdx.x;
  __shared__ float s_q[512];
  __shared__ float s_sc[128];
  __shared__ float s_red[64];
  __shared__ float s_bcast;

  for (int i = tid; i < 512; i += 256) s_q[i] = q[(size_t)b * 512 + i];
  __syncthreads();

  if (tid < 128) {
    const float* kp = k_enc + ((size_t)b * 128 + tid) * 512;
    float acc = 0.0f;
    for (int h = 0; h < 512; ++h) acc += s_q[h] * kp[h];
    if (mask[b * 128 + tid] == 0) acc = -1e9f;
    s_sc[tid] = acc;
  }
  __syncthreads();

  // softmax over 128
  if (tid < 64) s_red[tid] = fmaxf(s_sc[tid], s_sc[tid + 64]);
  __syncthreads();
  for (int off = 32; off > 0; off >>= 1) {
    if (tid < off) s_red[tid] = fmaxf(s_red[tid], s_red[tid + off]);
    __syncthreads();
  }
  if (tid == 0) s_bcast = s_red[0];
  __syncthreads();
  const float mx = s_bcast;
  __syncthreads();
  if (tid < 128) s_sc[tid] = __expf(s_sc[tid] - mx);
  __syncthreads();
  if (tid < 64) s_red[tid] = s_sc[tid] + s_sc[tid + 64];
  __syncthreads();
  for (int off = 32; off > 0; off >>= 1) {
    if (tid < off) s_red[tid] += s_red[tid + off];
    __syncthreads();
  }
  if (tid == 0) s_bcast = 1.0f / s_red[0];
  __syncthreads();
  const float inv = s_bcast;
  if (tid < 128) s_sc[tid] *= inv;
  __syncthreads();

  // context[d] = sum_s attn[s] * enc[b,s,d]  (fp32 inputs)
  unsigned short* xrow = x_cat + (size_t)b * 1792;
  unsigned short* crow = concat_bf + ((size_t)t * 32 + b) * 1792;
  for (int d = tid; d < 1024; d += 256) {
    const float* ep = enc + (size_t)b * 128 * 1024 + d;
    float acc = 0.0f;
    for (int s = 0; s < 128; ++s) acc += s_sc[s] * ep[(size_t)s * 1024];
    const unsigned short cb = f2bf(acc);
    xrow[256 + d] = cb;   // x = [emb | context]
    crow[512 + d] = cb;   // concat = [h_new | context | emb]
  }

  // embedding lookup (token t for batch b)
  const int tok = tgt[b * 64 + t];
  {
    const float e = emb_table[(size_t)tok * 256 + tid];
    const unsigned short eb = f2bf(e);
    xrow[tid] = eb;
    crow[1536 + tid] = eb;
  }
}

// ---------------------------------------------------------------------------
// GRU pointwise: r,z,n gates; h_new; update fp32 state + bf16 copies.
// ---------------------------------------------------------------------------
__global__ void gru_pointwise(const float* __restrict__ gi,
                              const float* __restrict__ gh,
                              float* __restrict__ h_state,
                              unsigned short* __restrict__ x_cat,
                              unsigned short* __restrict__ concat_bf,
                              int t) {
  int idx = blockIdx.x * blockDim.x + threadIdx.x;   // 0..16383
  if (idx >= 32 * 512) return;
  const int b = idx >> 9, j = idx & 511;
  const size_t gb = (size_t)b * 1536;
  const float ir  = gi[gb + j],         hr = gh[gb + j];
  const float iz  = gi[gb + 512 + j],   hz = gh[gb + 512 + j];
  const float in_ = gi[gb + 1024 + j],  hn = gh[gb + 1024 + j];
  const float r = 1.0f / (1.0f + __expf(-(ir + hr)));
  const float z = 1.0f / (1.0f + __expf(-(iz + hz)));
  const float n = tanhf(in_ + r * hn);
  const float hprev = h_state[idx];
  const float hnew = (1.0f - z) * n + z * hprev;
  h_state[idx] = hnew;
  const unsigned short hb = f2bf(hnew);
  x_cat[(size_t)b * 1792 + 1280 + j] = hb;            // h input for step t+1
  concat_bf[((size_t)t * 32 + b) * 1792 + j] = hb;    // h_new slot for logits
}

// ---------------------------------------------------------------------------
extern "C" void kernel_launch(void* const* d_in, const int* in_sizes, int n_in,
                              void* d_out, int out_size, void* d_ws, size_t ws_size,
                              hipStream_t stream) {
  (void)in_sizes; (void)n_in; (void)out_size; (void)ws_size;
  const int*   tgt    = (const int*)d_in[0];
  const int*   maskp  = (const int*)d_in[1];
  const float* enc    = (const float*)d_in[2];
  const float* h0     = (const float*)d_in[3];
  const float* embT   = (const float*)d_in[4];
  const float* Wk     = (const float*)d_in[5];
  const float* bk     = (const float*)d_in[6];
  const float* Wq     = (const float*)d_in[7];
  const float* bq     = (const float*)d_in[8];
  const float* W_ih   = (const float*)d_in[9];
  const float* W_hh   = (const float*)d_in[10];
  const float* b_ih   = (const float*)d_in[11];
  const float* b_hh   = (const float*)d_in[12];
  const float* W_out  = (const float*)d_in[13];
  const float* b_out  = (const float*)d_in[14];
  float* out = (float*)d_out;

  // workspace carve-up (256B aligned)
  char* wsp = (char*)d_ws;
  auto alloc = [&](size_t bytes) -> char* {
    char* p = wsp; wsp += (bytes + 255) & ~(size_t)255; return p;
  };
  unsigned short* Wout_bf = (unsigned short*)alloc((size_t)32000 * 1792 * 2); // 114.7 MB
  unsigned short* Wk_bf   = (unsigned short*)alloc((size_t)512 * 1024 * 2);
  unsigned short* Wq_bf   = (unsigned short*)alloc((size_t)512 * 512 * 2);
  unsigned short* Wih_bf  = (unsigned short*)alloc((size_t)1536 * 1280 * 2);
  unsigned short* Whh_bf  = (unsigned short*)alloc((size_t)1536 * 512 * 2);
  unsigned short* enc_bf  = (unsigned short*)alloc((size_t)32 * 128 * 1024 * 2);
  float*          k_enc   = (float*)alloc((size_t)32 * 128 * 512 * 4);
  unsigned short* cat_bf  = (unsigned short*)alloc((size_t)2048 * 1792 * 2);  // padded to 2048 rows
  unsigned short* x_cat   = (unsigned short*)alloc((size_t)32 * 1792 * 2);
  float*          qbuf    = (float*)alloc((size_t)32 * 512 * 4);
  float*          gi      = (float*)alloc((size_t)32 * 1536 * 4);
  float*          gh      = (float*)alloc((size_t)32 * 1536 * 4);
  float*          h_state = (float*)alloc((size_t)32 * 512 * 4);

  // 1) convert weights / encoder outputs to bf16
  convert_f32_bf16<<<2048, 256, 0, stream>>>(W_out, Wout_bf, (long)32000 * 1792);
  convert_f32_bf16<<<512,  256, 0, stream>>>(Wk,    Wk_bf,   (long)512 * 1024);
  convert_f32_bf16<<<512,  256, 0, stream>>>(Wq,    Wq_bf,   (long)512 * 512);
  convert_f32_bf16<<<1024, 256, 0, stream>>>(W_ih,  Wih_bf,  (long)1536 * 1280);
  convert_f32_bf16<<<512,  256, 0, stream>>>(W_hh,  Whh_bf,  (long)1536 * 512);
  convert_f32_bf16<<<1024, 256, 0, stream>>>(enc,   enc_bf,  (long)32 * 128 * 1024);

  // 2) k_enc = enc @ Wk^T + bk : M=4096, N=512, K=1024 (big-tile pipelined WMMA)
  gemm_bf16_wmma_big<<<dim3(2, 64), 256, 0, stream>>>(enc_bf, 1024, Wk_bf, 1024,
                                                      bk, k_enc, 512, 1024, 4096, 0);

  // 3) init recurrent state
  init_state<<<64, 256, 0, stream>>>(h0, h_state, x_cat);

  // 4) sequential recurrence, 63 steps (~6% of total FLOPs, latency-bound)
  for (int t = 0; t < 63; ++t) {
    // q = h @ Wq^T + bq : M=32, N=512, K=512
    gemm_bf16_wmma<<<dim3(2, 1), 256, 0, stream>>>(x_cat + 1280, 1792, Wq_bf, 512,
                                                   bq, qbuf, 512, 512);
    // scores/softmax/context + embedding
    attn_kernel<<<32, 256, 0, stream>>>(tgt, maskp, enc, embT, qbuf, k_enc,
                                        x_cat, cat_bf, t);
    // gi = [emb|ctx] @ W_ih^T + b_ih : M=32, N=1536, K=1280
    gemm_bf16_wmma<<<dim3(6, 1), 256, 0, stream>>>(x_cat, 1792, Wih_bf, 1280,
                                                   b_ih, gi, 1536, 1280);
    // gh = h @ W_hh^T + b_hh : M=32, N=1536, K=512
    gemm_bf16_wmma<<<dim3(6, 1), 256, 0, stream>>>(x_cat + 1280, 1792, Whh_bf, 512,
                                                   b_hh, gh, 1536, 512);
    // gates -> h_new
    gru_pointwise<<<64, 256, 0, stream>>>(gi, gh, h_state, x_cat, cat_bf, t);
  }

  // 5) logits = concat @ W_out^T + b_out : M=2016 (padded 2048), N=32000, K=1792
  //    out permuted to [B=32, T=63, V] layout; stores guarded at Mvalid=2016.
  gemm_bf16_wmma_big<<<dim3(125, 32), 256, 0, stream>>>(cat_bf, 1792, Wout_bf, 1792,
                                                        b_out, out, 32000, 1792, 2016, 1);
}
